// TextGATConv_37434934952835
// MI455X (gfx1250) — matrix operand order repaired and checked by reading, not compiled, not run on hardware
//
#include <hip/hip_runtime.h>
#include <hip/hip_bf16.h>

// Problem constants (match reference)
#define BB 32
#define LL 100
#define DD 128
#define HH 8
#define EE 64
#define CC 16
#define LP 128   // padded L (j dimension) to a multiple of 32 for WMMA K-steps
#define NEG_SLOPE 0.2f
#define MASK_VALF (-1e9f)
#define LN_EPS 1e-5f

typedef __attribute__((ext_vector_type(16))) _Float16 v16h;
typedef __attribute__((ext_vector_type(8)))  _Float16 v8h;
typedef __attribute__((ext_vector_type(8)))  float    v8f;
typedef __attribute__((ext_vector_type(4)))  float    v4f;

static __device__ __forceinline__ v8f wmma16(v16h a, v16h b, v8f c) {
  // D = A(16x32 f16) * B(32x16 f16) + C(16x16 f32)
  return __builtin_amdgcn_wmma_f32_16x16x32_f16(false, a, false, b, (short)0, c,
                                                false, false);
}

// ---- cross-lane helpers via ds_swizzle (group-of-32 XOR mode, imm-encoded:
// offset[14:10]=xor_mask, offset[4:0]=and_mask). One instruction, no address
// setup VALU (vs ds_bpermute lowering of __shfl_xor).
template <int IMM>
static __device__ __forceinline__ float swz(float v) {
  return __int_as_float(__builtin_amdgcn_ds_swizzle(__float_as_int(v), IMM));
}
#define SWZ_X1  0x041F
#define SWZ_X2  0x081F
#define SWZ_X4  0x101F
#define SWZ_X8  0x201F
#define SWZ_X16 0x401F
// sum-reduce across the 16 lanes of each half-wave
static __device__ __forceinline__ float red_half(float v) {
  v += swz<SWZ_X1>(v);
  v += swz<SWZ_X2>(v);
  v += swz<SWZ_X4>(v);
  v += swz<SWZ_X8>(v);
  return v;
}

static __device__ __forceinline__ v16h zero_a() {
  v16h z;
#pragma unroll
  for (int i = 0; i < 16; ++i) z[i] = (_Float16)0.0f;
  return z;
}
static __device__ __forceinline__ v8f zero_c() {
  v8f z;
#pragma unroll
  for (int i = 0; i < 8; ++i) z[i] = 0.0f;
  return z;
}
static __device__ __forceinline__ v16h cat8(v8h lo, v8h hi) {
  return __builtin_shufflevector(lo, hi, 0, 1, 2, 3, 4, 5, 6, 7, 8, 9, 10, 11,
                                 12, 13, 14, 15);
}
// A fragment from f32 row: halfs 0..7 = p[0..7], halfs 8..15 = p[16..23]
static __device__ __forceinline__ v16h loadA_f32(const float* p) {
  v4f a = *(const v4f*)(p);
  v4f b = *(const v4f*)(p + 4);
  v4f c = *(const v4f*)(p + 16);
  v4f d = *(const v4f*)(p + 20);
  v16h r;
#pragma unroll
  for (int i = 0; i < 4; ++i) {
    r[i]      = (_Float16)a[i];
    r[i + 4]  = (_Float16)b[i];
    r[i + 8]  = (_Float16)c[i];
    r[i + 12] = (_Float16)d[i];
  }
  return r;
}
// Same, but streaming (non-temporal): used for `e`, which is read exactly once.
static __device__ __forceinline__ v16h loadA_f32_nt(const float* p) {
  v4f a = __builtin_nontemporal_load((const v4f*)(p));
  v4f b = __builtin_nontemporal_load((const v4f*)(p + 4));
  v4f c = __builtin_nontemporal_load((const v4f*)(p + 16));
  v4f d = __builtin_nontemporal_load((const v4f*)(p + 20));
  v16h r;
#pragma unroll
  for (int i = 0; i < 4; ++i) {
    r[i]      = (_Float16)a[i];
    r[i + 4]  = (_Float16)b[i];
    r[i + 8]  = (_Float16)c[i];
    r[i + 12] = (_Float16)d[i];
  }
  return r;
}
// B fragment from f32 row: halfs 0..7 = p[0..7], halfs 8..15 = p[16..23]
static __device__ __forceinline__ v16h loadB_f32(const float* p) {
  v4f a = *(const v4f*)(p);
  v4f b = *(const v4f*)(p + 4);
  v4f c = *(const v4f*)(p + 16);
  v4f d = *(const v4f*)(p + 20);
  v16h r;
#pragma unroll
  for (int i = 0; i < 4; ++i) {
    r[i]      = (_Float16)a[i];
    r[i + 4]  = (_Float16)b[i];
    r[i + 8]  = (_Float16)c[i];
    r[i + 12] = (_Float16)d[i];
  }
  return r;
}
static __device__ __forceinline__ v16h loadB_f16(const _Float16* p) {
  v8h lo = *(const v8h*)(p);
  v8h hi = *(const v8h*)(p + 16);
  return cat8(lo, hi);
}
static __device__ __forceinline__ v16h loadA_f16(const _Float16* p) {
  v8h lo = *(const v8h*)(p);
  v8h hi = *(const v8h*)(p + 16);
  return cat8(lo, hi);
}

// ---------------------------------------------------------------------------
// Kernel 1: fold att_e into We:  v[h][e] = sum_c att_e[h,c] * We[h*C+c, e]
// Stored f16, 16 rows (h 8..15 zero-padded) x 64, row-major (WMMA B source).
// ---------------------------------------------------------------------------
__global__ void k_prep(const float* __restrict__ We,
                       const float* __restrict__ att_e,
                       _Float16* __restrict__ vf16) {
  int t = blockIdx.x * blockDim.x + threadIdx.x;
  if (t >= 16 * EE) return;
  int h = t >> 6, ee = t & 63;
  float s = 0.0f;
  if (h < HH) {
#pragma unroll
    for (int c = 0; c < CC; ++c)
      s += att_e[h * CC + c] * We[(size_t)(h * CC + c) * EE + ee];
  }
  vf16[t] = (_Float16)s;
}

// ---------------------------------------------------------------------------
// Kernel 2: xp = x @ Wx^T (WMMA), fused xi/xj head scores, store xpT f16.
// grid: (b, i-tile 0..7) = 256 blocks, 1 wave each.
// ---------------------------------------------------------------------------
__global__ void __launch_bounds__(32)
k_proj(const float* __restrict__ x, const float* __restrict__ Wx,
       const float* __restrict__ att_src, const float* __restrict__ att_dst,
       _Float16* __restrict__ xpT, float* __restrict__ xi,
       float* __restrict__ xj) {
  const int b    = blockIdx.x >> 3;
  const int i0   = (blockIdx.x & 7) * 16;
  const int lid  = threadIdx.x;
  const int ln   = lid & 15;
  const int half = lid >> 4;
  const int oA   = half * 8;
  const int oB   = half * 16;

  const int irow = i0 + ln;
  const bool iv  = irow < LL;
  const float* xr = x + (size_t)(b * LL + (iv ? irow : 0)) * DD;
  v16h at[4];
#pragma unroll
  for (int ks = 0; ks < 4; ++ks)
    at[ks] = iv ? loadA_f32(xr + ks * 32 + oA) : zero_a();

#pragma unroll
  for (int n = 0; n < HH; ++n) {  // n-tile == head n (C == 16)
    v8f acc = zero_c();
#pragma unroll
    for (int ks = 0; ks < 4; ++ks) {
      const float* wr = Wx + (size_t)(n * 16 + ln) * DD + ks * 32 + oA;
      acc = wmma16(at[ks], loadB_f32(wr), acc);
    }
    // transposed f16 store: xpT[b][n*16+ln][i0+half*8 + r] (contiguous in r)
    v8h pk;
#pragma unroll
    for (int r = 0; r < 8; ++r) pk[r] = (_Float16)acc[r];
    *(v8h*)(xpT + ((size_t)(b * DD) + n * 16 + ln) * LP + i0 + half * 8) = pk;

    // per-head attention scores: reduce over C (16 lanes of each half-wave)
    const float wd = att_dst[n * CC + ln];
    const float ws = att_src[n * CC + ln];
    float pd[8], ps[8];
#pragma unroll
    for (int r = 0; r < 8; ++r) {
      pd[r] = red_half(acc[r] * wd);
      ps[r] = red_half(acc[r] * ws);
    }
    if (ln == 0) {
#pragma unroll
      for (int r = 0; r < 8; ++r) {
        const int ir = i0 + half * 8 + r;  // < 128, buffer is padded
        xi[((size_t)(b * LP) + ir) * HH + n] = pd[r];
        xj[((size_t)(b * LP) + ir) * HH + n] = ps[r];
      }
    }
  }
}

// ---------------------------------------------------------------------------
// Kernel 3: fused edge-score GEMM + softmax + aggregation GEMM.
// grid: (b, i-tile 0..6) = 224 workgroups, 8 waves (256 threads).
// Phase 1: wave w handles dest rows {2w, 2w+1}: es via WMMA (M=j, N=h, K=E),
//          add xi+xj, leaky-relu, adj mask, softmax over j in registers,
//          alpha -> LDS in WMMA-A layout (packed b128 stores).
// Phase 2: wave w = head w: out = alpha @ xp_h via WMMA (M=i, K=j, N=C).
// ---------------------------------------------------------------------------
__global__ void __launch_bounds__(256)
k_attn(const float* __restrict__ e, const int* __restrict__ adj,
       const _Float16* __restrict__ vf16, const _Float16* __restrict__ xpT,
       const float* __restrict__ xi, const float* __restrict__ xj,
       float* __restrict__ agg) {
  __shared__ __align__(16) _Float16 alpha_s[16 * HH * LP];  // 32 KB
  __shared__ int   adj_s[16 * LP];                          // 8 KB
  __shared__ float xj_s[LP * HH];                           // 4 KB
  __shared__ float xi_s[16 * HH];

  const int b    = blockIdx.x / 7;
  const int i0   = (blockIdx.x % 7) * 16;
  const int tid  = threadIdx.x;
  const int lid  = tid & 31;
  const int w    = tid >> 5;
  const int ln   = lid & 15;
  const int half = lid >> 4;
  const int oA   = half * 8;
  const int oB   = half * 16;

  // cooperative staging
  for (int q = tid; q < 16 * LP; q += 256) {
    const int r = q >> 7, j = q & 127;
    const int i = i0 + r;
    adj_s[q] = (i < LL && j < LL) ? adj[(size_t)(b * LL + i) * LL + j] : 0;
  }
  for (int q = tid; q < LP * HH; q += 256)
    xj_s[q] = xj[(size_t)b * LP * HH + q];
  if (tid < 16 * HH) xi_s[tid] = xi[((size_t)(b * LP) + i0) * HH + tid];
  __syncthreads();

  // B fragments for edge projection (v is 16x64 f16 row-major)
  const _Float16* vp = vf16 + ln * EE;
  const v16h bv0 = loadB_f16(vp + 0 + oA);
  const v16h bv1 = loadB_f16(vp + 32 + oA);

  // ---- phase 1 ----
  for (int rr = 0; rr < 2; ++rr) {
    const int il = w * 2 + rr;
    const int i  = i0 + il;
    const bool iv = i < LL;

    // prefetch next destination row's edge block (global_prefetch_b8)
    if (rr == 0) {
      const int ip = i0 + il + 1;
      if (ip < LL) {
        const float* pf =
            e + ((size_t)(b * LL + ip) * LL + (ln << 3)) * EE + oA * 16;
        __builtin_prefetch(pf, 0, 1);
      }
    }

    float sv[8][8];
#pragma unroll
    for (int t = 0; t < 8; ++t) {
      const int jr = t * 16 + ln;                 // A row = source node j
      const bool valid = iv && (jr < LL);
      const float* ap =
          e + ((size_t)(b * LL + (iv ? i : 0)) * LL + (valid ? jr : 0)) * EE;
      v16h a0 = valid ? loadA_f32_nt(ap + oA)      : zero_a();
      v16h a1 = valid ? loadA_f32_nt(ap + 32 + oA) : zero_a();
      v8f acc = zero_c();
      acc = wmma16(a0, bv0, acc);   // K = 0..31
      acc = wmma16(a1, bv1, acc);   // K = 32..63
      const float xiv = xi_s[il * HH + (ln & 7)];
#pragma unroll
      for (int r = 0; r < 8; ++r) {
        const int j = t * 16 + half * 8 + r;
        float s = acc[r] + xiv + xj_s[j * HH + (ln & 7)];
        s = s > 0.0f ? s : NEG_SLOPE * s;
        if (adj_s[il * LP + j] == 0) s = MASK_VALF;
        sv[t][r] = s;
      }
    }
    // softmax over j: per-lane reduce + join the two j-halves (lane ^ 16)
    float mx = -3.0e38f;
#pragma unroll
    for (int t = 0; t < 8; ++t)
#pragma unroll
      for (int r = 0; r < 8; ++r) mx = fmaxf(mx, sv[t][r]);
    mx = fmaxf(mx, swz<SWZ_X16>(mx));
    float sum = 0.0f;
#pragma unroll
    for (int t = 0; t < 8; ++t)
#pragma unroll
      for (int r = 0; r < 8; ++r) {
        const float ev = __expf(sv[t][r] - mx);
        sv[t][r] = ev;
        sum += ev;
      }
    sum += swz<SWZ_X16>(sum);
    const float inv = 1.0f / sum;
    if (ln < HH) {  // valid heads only
#pragma unroll
      for (int t = 0; t < 8; ++t) {
        v8h pk;
#pragma unroll
        for (int r = 0; r < 8; ++r) pk[r] = (_Float16)(sv[t][r] * inv);
        *(v8h*)(alpha_s + (il * HH + ln) * LP + t * 16 + half * 8) = pk;
      }
    }
  }
  __syncthreads();

  // ---- phase 2: wave w == head w ----
  const int hh = w;
  const _Float16* bp   = xpT + ((size_t)(b * DD) + hh * CC + ln) * LP;
  const _Float16* arow = alpha_s + (ln * HH + hh) * LP;
  v8f acc = zero_c();
#pragma unroll
  for (int ks = 0; ks < 4; ++ks) {
    v16h bt = loadB_f16(bp + ks * 32 + oA);
    v16h at = loadA_f16(arow + ks * 32 + oA);
    acc = wmma16(at, bt, acc);
  }
#pragma unroll
  for (int r = 0; r < 8; ++r) {
    const int ir = i0 + half * 8 + r;
    if (ir < LL)
      agg[((size_t)(b * LL) + ir) * DD + hh * CC + ln] = acc[r];
  }
}

// ---------------------------------------------------------------------------
// Kernel 4: h = agg @ ffn_w^T + b, fused LayerNorm + ReLU.
// grid: 3200/16 = 200 blocks, 1 wave each.
// ---------------------------------------------------------------------------
__global__ void __launch_bounds__(32)
k_ffn(const float* __restrict__ agg, const float* __restrict__ ffn_w,
      const float* __restrict__ ffn_b, const float* __restrict__ ln_g,
      const float* __restrict__ ln_b, float* __restrict__ out) {
  const int m0   = blockIdx.x * 16;
  const int lid  = threadIdx.x;
  const int ln   = lid & 15;
  const int half = lid >> 4;
  const int oA   = half * 8;
  const int oB   = half * 16;

  const float* ar = agg + (size_t)(m0 + ln) * DD;
  v16h at[4];
#pragma unroll
  for (int ks = 0; ks < 4; ++ks) at[ks] = loadA_f32(ar + ks * 32 + oA);

  float dt[8][8];
#pragma unroll
  for (int n = 0; n < 8; ++n) {
    v8f acc = zero_c();
#pragma unroll
    for (int ks = 0; ks < 4; ++ks)
      acc = wmma16(at[ks],
                   loadB_f32(ffn_w + (size_t)(n * 16 + ln) * DD + ks * 32 + oA),
                   acc);
    const float bb = ffn_b[n * 16 + ln];
#pragma unroll
    for (int r = 0; r < 8; ++r) dt[n][r] = acc[r] + bb;
  }

  // LayerNorm per row (row = m0 + half*8 + r): reduce over 8 tiles x 16 lanes
  float mean[8], rsn[8];
#pragma unroll
  for (int r = 0; r < 8; ++r) {
    float s = 0.0f;
#pragma unroll
    for (int n = 0; n < 8; ++n) s += dt[n][r];
    mean[r] = red_half(s) * (1.0f / DD);
  }
#pragma unroll
  for (int r = 0; r < 8; ++r) {
    float s = 0.0f;
#pragma unroll
    for (int n = 0; n < 8; ++n) {
      const float d = dt[n][r] - mean[r];
      s += d * d;
    }
    rsn[r] = rsqrtf(red_half(s) * (1.0f / DD) + LN_EPS);
  }

#pragma unroll
  for (int n = 0; n < 8; ++n) {
    const int col = n * 16 + ln;
    const float g = ln_g[col];
    const float bt = ln_b[col];
#pragma unroll
    for (int r = 0; r < 8; ++r) {
      float y = (dt[n][r] - mean[r]) * rsn[r] * g + bt;
      y = y > 0.0f ? y : 0.0f;
      out[(size_t)(m0 + half * 8 + r) * DD + col] = y;
    }
  }
}

// ---------------------------------------------------------------------------
extern "C" void kernel_launch(void* const* d_in, const int* in_sizes, int n_in,
                              void* d_out, int out_size, void* d_ws,
                              size_t ws_size, hipStream_t stream) {
  (void)in_sizes; (void)n_in; (void)out_size; (void)ws_size;
  const float* x       = (const float*)d_in[0];
  const int*   adj     = (const int*)d_in[1];
  const float* e       = (const float*)d_in[2];
  const float* Wx      = (const float*)d_in[3];
  const float* We      = (const float*)d_in[4];
  const float* att_src = (const float*)d_in[5];
  const float* att_dst = (const float*)d_in[6];
  const float* att_e   = (const float*)d_in[7];
  const float* ffn_w   = (const float*)d_in[8];
  const float* ffn_b   = (const float*)d_in[9];
  const float* ln_g    = (const float*)d_in[10];
  const float* ln_b    = (const float*)d_in[11];
  float* out = (float*)d_out;

  // workspace carve-up (~2.9 MB total)
  char* ws = (char*)d_ws;
  _Float16* vf16 = (_Float16*)ws;                                   // 2 KB
  _Float16* xpT  = (_Float16*)(ws + 2048);                          // 1 MB
  float*    xi   = (float*)(ws + 2048 + (size_t)BB * DD * LP * 2);  // 128 KB
  float*    xj   = xi + (size_t)BB * LP * HH;                       // 128 KB
  float*    agg  = xj + (size_t)BB * LP * HH;                       // 1.6 MB

  k_prep<<<4, 256, 0, stream>>>(We, att_e, vf16);
  k_proj<<<BB * 8, 32, 0, stream>>>(x, Wx, att_src, att_dst, xpT, xi, xj);
  k_attn<<<BB * 7, 256, 0, stream>>>(e, adj, vf16, xpT, xi, xj, agg);
  k_ffn<<<(BB * LL) / 16, 32, 0, stream>>>(agg, ffn_w, ffn_b, ln_g, ln_b, out);
}
